// NimbusLinear_62362925138767
// MI455X (gfx1250) — matrix operand order repaired and checked by reading, not compiled, not run on hardware
//
#include <hip/hip_runtime.h>
#include <hip/hip_bf16.h>

// ---------------------------------------------------------------------------
// NimbusLinear forward for MI455X (gfx1250, wave32, WMMA).
//
//   idx[n,c]  = depth-4 tree traversal over 4 gathered features
//   out[n,o]  = sum_c lut[o, c, idx[n,c]]
//             = (one-hot E [N x 4096]) @ (lut reshaped [4096 x 4096])^T
//
// Decode runs on v_wmma_f32_16x16x32_f16: A = exact one-hot in f16,
// B = lut in f16, C/D = f32. 137 GFLOP of matrix work, L2-resident B.
// ---------------------------------------------------------------------------

typedef __attribute__((ext_vector_type(16))) _Float16 v16h;
typedef __attribute__((ext_vector_type(4)))  _Float16 v4h;
typedef __attribute__((ext_vector_type(8)))  _Float16 v8h;
typedef __attribute__((ext_vector_type(4)))  float    v4f;
typedef __attribute__((ext_vector_type(8)))  float    v8f;

#define N_ROWS 4096
#define IN_F   4096
#define OUT_F  4096
#define CBOOK  256
#define KLEAF  16
#define KTOT   (CBOOK * KLEAF)   // 4096

// ---------------------------------------------------------------------------
// Kernel 1: tree encode.  One block per input row n, one thread per codebook c.
// node i at depth d uses feature dims[4c+d]; go right iff x > thresholds[15c+i].
// ---------------------------------------------------------------------------
__global__ __launch_bounds__(256) void nimbus_encode(
    const float* __restrict__ in, const int* __restrict__ dims,
    const float* __restrict__ thr, unsigned char* __restrict__ idx) {
  __shared__ int   s_dims[CBOOK * 4];
  __shared__ float s_thr[CBOOK * 15];
  for (int i = threadIdx.x; i < CBOOK * 4; i += 256)  s_dims[i] = dims[i];
  for (int i = threadIdx.x; i < CBOOK * 15; i += 256) s_thr[i] = thr[i];
  __syncthreads();

  const int c = threadIdx.x;
  const int n = blockIdx.x;
  const float* __restrict__ row = in + (size_t)n * IN_F;

  int node = 0;
#pragma unroll
  for (int d = 0; d < 4; ++d) {
    float x = row[s_dims[c * 4 + d]];
    float t = s_thr[c * 15 + node];
    node = 2 * node + 1 + ((x > t) ? 1 : 0);
  }
  idx[(size_t)n * CBOOK + c] = (unsigned char)(node - 15);  // leaf 0..15
}

// ---------------------------------------------------------------------------
// Kernel 2: lut f32 -> f16.  lut layout (o, c, k) is already B-column-major
// (each o-row is a contiguous 4096-length K vector), so a plain cast suffices.
// ---------------------------------------------------------------------------
__global__ __launch_bounds__(256) void nimbus_lutcvt(
    const float* __restrict__ lut, _Float16* __restrict__ bt) {
  size_t i = ((size_t)blockIdx.x * 256 + threadIdx.x);  // one float4 per thread
  v4f x = ((const v4f*)lut)[i];
  ((v4h*)bt)[i] = __builtin_convertvector(x, v4h);
}

// ---------------------------------------------------------------------------
// Kernel 3: WMMA LUT-GEMM.
// Block = 8 waves. Wave tile = 32n x 64o, K loop of 4096 in steps of 32
// (2 codebooks / step).  Per step: 2 one-hot A tiles (built from 2x u16 idx
// loads + unrolled selects), 4 B tiles (two 16B loads per lane), and
// 8 v_wmma_f32_16x16x32_f16.
//
// A layout (ISA 16-bit A 16x32): lanes 0-15 = row M, halves {K0..7, K16..23};
// lanes 16-31 = row M, halves {K8..15, K24..31}.
// B layout (ISA 16-bit B):       lanes 0-15 = col N, halves {K0..7, K16..23};
// lanes 16-31 = col N, halves {K8..15, K24..31}.
// D layout: lane<16 -> col N=lane, VGPR r -> row M=r; lane>=16 -> M=r+8.
// ---------------------------------------------------------------------------
__global__ __launch_bounds__(256) void nimbus_lutmm(
    const unsigned char* __restrict__ idx,
    const _Float16* __restrict__ bt,    // [OUT_F][KTOT] f16
    float* __restrict__ out) {          // [N_ROWS][OUT_F] f32
  const int lane   = threadIdx.x & 31;
  const int wave   = threadIdx.x >> 5;       // 0..7
  const int laneLo = lane & 15;
  const int hi     = lane >> 4;              // 0 or 1

  const int o0 = blockIdx.x * 512 + wave * 64;  // 8 o-blocks of 512
  const int n0 = blockIdx.y * 32;               // 128 n-blocks of 32

  v8f acc[2][4];
#pragma unroll
  for (int p = 0; p < 2; ++p)
#pragma unroll
    for (int t = 0; t < 4; ++t) acc[p][t] = (v8f)0.0f;

  const unsigned char* __restrict__ idxRow0 = idx + (size_t)(n0 + laneLo) * CBOOK;
  const unsigned char* __restrict__ idxRow1 = idxRow0 + (size_t)16 * CBOOK;

  const _Float16* __restrict__ brow0 = bt + (size_t)(o0 +  0 + laneLo) * KTOT;
  const _Float16* __restrict__ brow1 = bt + (size_t)(o0 + 16 + laneLo) * KTOT;
  const _Float16* __restrict__ brow2 = bt + (size_t)(o0 + 32 + laneLo) * KTOT;
  const _Float16* __restrict__ brow3 = bt + (size_t)(o0 + 48 + laneLo) * KTOT;

  const int       sShift = hi ? 8 : 0;
  const _Float16  ONE  = (_Float16)1.0f;
  const _Float16  ZERO = (_Float16)0.0f;

  for (int ks = 0; ks < KTOT; ks += 32) {
    const int c0 = ks >> 4;                         // even codebook index
    const unsigned int pr0 = *(const unsigned short*)(idxRow0 + c0);
    const unsigned int pr1 = *(const unsigned short*)(idxRow1 + c0);
    const int s0a = (int)(pr0 & 0xFFu) - sShift;    // codebook c0, A tile 0
    const int s1a = (int)(pr0 >> 8)    - sShift;    // codebook c0+1, A tile 0
    const int s0b = (int)(pr1 & 0xFFu) - sShift;    // codebook c0, A tile 1
    const int s1b = (int)(pr1 >> 8)    - sShift;    // codebook c0+1, A tile 1

    v16h a0, a1;
#pragma unroll
    for (int j = 0; j < 8; ++j) {
      a0[j]     = (j == s0a) ? ONE : ZERO;
      a0[j + 8] = (j == s1a) ? ONE : ZERO;
      a1[j]     = (j == s0b) ? ONE : ZERO;
      a1[j + 8] = (j == s1b) ? ONE : ZERO;
    }

    const int kb = ks + (hi ? 8 : 0);               // halves {kb..kb+7, kb+16..kb+23}
    v16h b0 = __builtin_shufflevector(*(const v8h*)(brow0 + kb), *(const v8h*)(brow0 + kb + 16), 0,1,2,3,4,5,6,7,8,9,10,11,12,13,14,15);
    v16h b1 = __builtin_shufflevector(*(const v8h*)(brow1 + kb), *(const v8h*)(brow1 + kb + 16), 0,1,2,3,4,5,6,7,8,9,10,11,12,13,14,15);
    v16h b2 = __builtin_shufflevector(*(const v8h*)(brow2 + kb), *(const v8h*)(brow2 + kb + 16), 0,1,2,3,4,5,6,7,8,9,10,11,12,13,14,15);
    v16h b3 = __builtin_shufflevector(*(const v8h*)(brow3 + kb), *(const v8h*)(brow3 + kb + 16), 0,1,2,3,4,5,6,7,8,9,10,11,12,13,14,15);

    acc[0][0] = __builtin_amdgcn_wmma_f32_16x16x32_f16(false, a0, false, b0, (short)0, acc[0][0], false, false);
    acc[1][0] = __builtin_amdgcn_wmma_f32_16x16x32_f16(false, a1, false, b0, (short)0, acc[1][0], false, false);
    acc[0][1] = __builtin_amdgcn_wmma_f32_16x16x32_f16(false, a0, false, b1, (short)0, acc[0][1], false, false);
    acc[1][1] = __builtin_amdgcn_wmma_f32_16x16x32_f16(false, a1, false, b1, (short)0, acc[1][1], false, false);
    acc[0][2] = __builtin_amdgcn_wmma_f32_16x16x32_f16(false, a0, false, b2, (short)0, acc[0][2], false, false);
    acc[1][2] = __builtin_amdgcn_wmma_f32_16x16x32_f16(false, a1, false, b2, (short)0, acc[1][2], false, false);
    acc[0][3] = __builtin_amdgcn_wmma_f32_16x16x32_f16(false, a0, false, b3, (short)0, acc[0][3], false, false);
    acc[1][3] = __builtin_amdgcn_wmma_f32_16x16x32_f16(false, a1, false, b3, (short)0, acc[1][3], false, false);
  }

  // Store D tiles: lane<16 -> rows r..r (M=r), lane>=16 -> M=r+8.
#pragma unroll
  for (int p = 0; p < 2; ++p) {
    const int nb = n0 + p * 16 + sShift;
#pragma unroll
    for (int t = 0; t < 4; ++t) {
      const int ocol = o0 + t * 16 + laneLo;
#pragma unroll
      for (int r = 0; r < 8; ++r) {
        out[(size_t)(nb + r) * OUT_F + ocol] = acc[p][t][r];
      }
    }
  }
}

// ---------------------------------------------------------------------------
// Launch.  d_in: 0=inputMatrix f32, 1=dims i32, 2=selection_matrix (unused),
// 3=thresholds f32, 4=tree_des_mat (unused), 5=lut f32.
// d_ws: [0, 32MB) lut f16, [32MB, 33MB) idx bytes.
// ---------------------------------------------------------------------------
extern "C" void kernel_launch(void* const* d_in, const int* in_sizes, int n_in,
                              void* d_out, int out_size, void* d_ws, size_t ws_size,
                              hipStream_t stream) {
  const float* in   = (const float*)d_in[0];
  const int*   dims = (const int*)d_in[1];
  const float* thr  = (const float*)d_in[3];
  const float* lut  = (const float*)d_in[5];
  float*       out  = (float*)d_out;

  _Float16*      bt   = (_Float16*)d_ws;
  unsigned char* idxb = (unsigned char*)d_ws + (size_t)OUT_F * KTOT * sizeof(_Float16);

  // 1) encode: 4096 rows x 256 codebooks
  nimbus_encode<<<N_ROWS, 256, 0, stream>>>(in, dims, thr, idxb);

  // 2) lut f32 -> f16: 16,777,216 elems, 4 per thread
  nimbus_lutcvt<<<(OUT_F * KTOT) / (256 * 4), 256, 0, stream>>>(lut, bt);

  // 3) WMMA LUT-GEMM: grid (o-blocks=8, n-blocks=128), 8 waves/block
  nimbus_lutmm<<<dim3(8, 128, 1), 256, 0, stream>>>(idxb, bt, out);
}
